// FeatureAlign_V2_67783173865736
// MI455X (gfx1250) — compile-verified, matrix operand order checked by reading, not run on hardware
//
#include <hip/hip_runtime.h>
#include <math.h>

// ---------------------------------------------------------------------------
// FeatureAlign V2 (FaPN) for gfx1250: all dense math on v_wmma_f32_16x16x32_f16
// B=8, C=128, H=W=64, DG=8, K=9 taps, cpg=16.
// ---------------------------------------------------------------------------

typedef __attribute__((ext_vector_type(16))) _Float16 v16h;
typedef __attribute__((ext_vector_type(8)))  float    v8f;

#define BB   8
#define CC   128
#define Hs   32
#define Ws   32
#define Hh   64
#define Wh   64
#define HW   4096        // 64*64
#define NCM  216         // DG*3*K = 8*27

union Frag { v16h v; unsigned u[8]; };

__device__ __forceinline__ float sigmoidf_(float x) { return 1.0f / (1.0f + __expf(-x)); }

// ---------------------------------------------------------------------------
// Generic 64(M) x 64(N) WMMA GEMM tile core.
// Block = 128 threads = 4 waves. Wave w owns M rows [16w, 16w+16); 4
// accumulators cover N=64. K consumed in chunks of 32 staged via LDS (f16,
// padded stride 34 halfs -> even-k pairs stay 4B aligned for 32-bit ds reads).
// Global fetches for the chunk are staged in REGISTERS before the barrier so
// they overlap the previous chunk's WMMAs, then dumped to LDS.
// fa(m,k): global A element (f32). fb(n,k): global B element (f32).
// fs(m,n,v): epilogue store.
// ---------------------------------------------------------------------------
template <class FA, class FB, class FS>
__device__ void gemm64x64(int K, FA fa, FB fb, FS fs) {
  __shared__ _Float16 Al[64 * 34];
  __shared__ _Float16 Bl[64 * 34];
  const int tid  = threadIdx.x;
  const int lane = tid & 31;
  const int wave = tid >> 5;
  v8f acc[4] = {{}, {}, {}, {}};

  for (int k0 = 0; k0 < K; k0 += 32) {
    // ---- stage chunk into registers (overlaps previous chunk's WMMAs) ----
    _Float16 ar[16], br[16];
#pragma unroll
    for (int i = 0; i < 16; ++i) {       // 2048 elements each for A and B
      int e = i * 128 + tid;
      ar[i] = (_Float16)fa(e >> 5, k0 + (e & 31));  // A: k fastest, coalesced
      br[i] = (_Float16)fb(e & 63, k0 + (e >> 6));  // B: n fastest, coalesced
    }
    __syncthreads();                     // LDS free of previous chunk's readers
#pragma unroll
    for (int i = 0; i < 16; ++i) {
      int e = i * 128 + tid;
      Al[(e >> 5) * 34 + (e & 31)] = ar[i];
      Bl[(e & 63) * 34 + (e >> 6)] = br[i];
    }
    __syncthreads();

    // ISA 16-bit A 16x32 layout: lanes 0-15 -> K {0..7,16..23}, lanes 16-31 -> +8.
    const int sel8 = (lane >> 4) * 8;
    Frag af;
    {
      const int row = wave * 16 + (lane & 15);
#pragma unroll
      for (int j = 0; j < 8; ++j) {
        int kk = ((j < 4) ? 2 * j : 2 * j + 8) + sel8;   // even -> aligned b32
        af.u[j] = *(const unsigned*)&Al[row * 34 + kk];
      }
    }
#pragma unroll
    for (int t = 0; t < 4; ++t) {
      Frag bf;
      const int col = t * 16 + (lane & 15);
#pragma unroll
      for (int j = 0; j < 8; ++j) {
        int kk = ((j < 4) ? 2 * j : 2 * j + 8) + sel8;
        bf.u[j] = *(const unsigned*)&Bl[col * 34 + kk];
      }
      acc[t] = __builtin_amdgcn_wmma_f32_16x16x32_f16(
          /*neg_a=*/false, af.v, /*neg_b=*/false, bf.v,
          /*c_mod=*/(short)0, acc[t], /*reuse_a=*/false, /*reuse_b=*/false);
    }
  }

  // C/D layout: lanes 0-15: N=lane, M=vgpr; lanes 16-31: N=lane-16, M=vgpr+8.
  const int mo = (lane >> 4) ? 8 : 0;
#pragma unroll
  for (int t = 0; t < 4; ++t)
#pragma unroll
    for (int r = 0; r < 8; ++r)
      fs(wave * 16 + mo + r, t * 16 + (lane & 15), acc[t][r]);
}

// ---------------------------------------------------------------------------
// Stage 1: bilinear 2x upsample (half-pixel centers) feat_s -> feat_up
// ---------------------------------------------------------------------------
__global__ __launch_bounds__(256) void upsample_kernel(const float* __restrict__ fsrc,
                                                       float* __restrict__ fu) {
  int idx = blockIdx.x * blockDim.x + threadIdx.x;
  if (idx >= BB * CC * HW) return;
  int x = idx & 63, y = (idx >> 6) & 63;
  int bc = idx >> 12;
  float sy = y * 0.5f - 0.25f, sx = x * 0.5f - 0.25f;
  int y0 = (int)floorf(sy), x0 = (int)floorf(sx);
  float fy = sy - y0, fx = sx - x0;
  int y0c = min(max(y0, 0), Hs - 1), y1c = min(max(y0 + 1, 0), Hs - 1);
  int x0c = min(max(x0, 0), Ws - 1), x1c = min(max(x0 + 1, 0), Ws - 1);
  const float* p = fsrc + (size_t)bc * (Hs * Ws);
  float v00 = p[y0c * Ws + x0c], v01 = p[y0c * Ws + x1c];
  float v10 = p[y1c * Ws + x0c], v11 = p[y1c * Ws + x1c];
  fu[idx] = (1.f - fy) * ((1.f - fx) * v00 + fx * v01) +
            fy * ((1.f - fx) * v10 + fx * v11);
}

// ---------------------------------------------------------------------------
// Stage 2: global-avg-pool + SE gate  -> scale[b,c] = 1 + sigmoid(gap @ w_atten^T)
// ---------------------------------------------------------------------------
__global__ __launch_bounds__(256) void gap_atten_kernel(const float* __restrict__ feat_l,
                                                        const float* __restrict__ w_atten,
                                                        float* __restrict__ scale) {
  __shared__ float g[CC];
  const int b = blockIdx.x;
  const int lane = threadIdx.x & 31, wave = threadIdx.x >> 5;
  for (int c = wave; c < CC; c += 8) {          // one wave per channel, coalesced
    const float* p = feat_l + ((size_t)b * CC + c) * HW;
    float s = 0.f;
    for (int i = lane; i < HW; i += 32) s += p[i];
    for (int off = 16; off; off >>= 1) s += __shfl_xor(s, off, 32);
    if (lane == 0) g[c] = s * (1.0f / HW);
  }
  __syncthreads();
  for (int o = threadIdx.x; o < CC; o += 256) {
    float d = 0.f;
    for (int c = 0; c < CC; ++c) d += g[c] * w_atten[o * CC + c];
    scale[b * CC + o] = 1.0f + sigmoidf_(d);
  }
}

// ---------------------------------------------------------------------------
// Stage 3: feat_arm = w_fsm @ (feat_l * scale)     M=128 K=128 N=4096
// ---------------------------------------------------------------------------
__global__ __launch_bounds__(128) void fsm_kernel(const float* __restrict__ feat_l,
                                                  const float* __restrict__ w_fsm,
                                                  const float* __restrict__ scale,
                                                  float* __restrict__ feat_arm) {
  const int n0 = blockIdx.x * 64, m0 = blockIdx.y * 64, b = blockIdx.z;
  gemm64x64(CC,
      [=](int m, int k) { return w_fsm[(m0 + m) * CC + k]; },
      [=](int n, int k) {
        return feat_l[((size_t)b * CC + k) * HW + n0 + n] * scale[b * CC + k];
      },
      [=](int m, int n, float v) {
        feat_arm[((size_t)b * CC + m0 + m) * HW + n0 + n] = v;
      });
}

// ---------------------------------------------------------------------------
// Stage 4: off_feat = w_off @ cat(feat_arm, 2*feat_up)   M=128 K=256 N=4096
// ---------------------------------------------------------------------------
__global__ __launch_bounds__(128) void off_kernel(const float* __restrict__ feat_arm,
                                                  const float* __restrict__ feat_up,
                                                  const float* __restrict__ w_off,
                                                  float* __restrict__ off_feat) {
  const int n0 = blockIdx.x * 64, m0 = blockIdx.y * 64, b = blockIdx.z;
  gemm64x64(2 * CC,
      [=](int m, int k) { return w_off[(m0 + m) * (2 * CC) + k]; },
      [=](int n, int k) {
        return (k < CC)
                   ? feat_arm[((size_t)b * CC + k) * HW + n0 + n]
                   : 2.0f * feat_up[((size_t)b * CC + (k - CC)) * HW + n0 + n];
      },
      [=](int m, int n, float v) {
        off_feat[((size_t)b * CC + m0 + m) * HW + n0 + n] = v;
      });
}

// ---------------------------------------------------------------------------
// Stage 5: om = conv3x3(off_feat, w_cm) + b_cm  (implicit GEMM, im2col in the
// B-fetch; K ordered tap-major so each 32-chunk stays within one tap).
// Epilogue applies sigmoid to mask channels (144..215).  M=216 K=1152 N=4096
// ---------------------------------------------------------------------------
__global__ __launch_bounds__(128) void cm_kernel(const float* __restrict__ off_feat,
                                                 const float* __restrict__ w_cm,
                                                 const float* __restrict__ b_cm,
                                                 float* __restrict__ om) {
  const int n0 = blockIdx.x * 64, m0 = blockIdx.y * 64, b = blockIdx.z;
  gemm64x64(9 * CC,
      [=](int m, int k) {
        int mg = m0 + m;
        if (mg >= NCM) return 0.0f;
        int c = k & 127, tap = k >> 7;
        return w_cm[((size_t)mg * CC + c) * 9 + tap];
      },
      [=](int n, int k) {
        int c = k & 127, tap = k >> 7;
        int hw = n0 + n;
        int y = (hw >> 6) + tap / 3 - 1;
        int x = (hw & 63) + tap % 3 - 1;
        if (y < 0 || y >= Hh || x < 0 || x >= Wh) return 0.0f;
        return off_feat[((size_t)b * CC + c) * HW + y * Wh + x];
      },
      [=](int m, int n, float v) {
        int mg = m0 + m;
        if (mg >= NCM) return;
        float r = v + b_cm[mg];
        if (mg >= 144) r = sigmoidf_(r);               // mask channels
        om[((size_t)b * NCM + mg) * HW + n0 + n] = r;  // off_y/off_x raw
      });
}

// ---------------------------------------------------------------------------
// Stage 6: deformable conv v2 as GEMM: out[o,hw] = sum_{g,c,tap} wr * val.
// Bilinear coefficients depend only on (g,tap,pixel) -> precompute the 72x64
// coefficient sets ONCE per N-tile into LDS (weights carry mask + border
// validity), amortizing the sampling math over the 16 channels of each group
// (~16x VALU reduction in the dominant stage). The B-fetch is then 3 DS reads
// + 4 clamped gathers + 4 FMAs.
// Epilogue: relu(acc + b_dcn) + feat_arm -> d_out.   M=128 K=1152 N=4096
// LDS: 72*64*4 f16 weights (36.9KB) + 72*64 packed pos (18.4KB) + GEMM tiles
// (8.7KB) = 64.0KB.
// ---------------------------------------------------------------------------
__global__ __launch_bounds__(128) void dcn_kernel(const float* __restrict__ feat_up,
                                                  const float* __restrict__ om,
                                                  const float* __restrict__ w_dcn,
                                                  const float* __restrict__ b_dcn,
                                                  const float* __restrict__ feat_arm,
                                                  float* __restrict__ out) {
  __shared__ _Float16 sWgt[72 * 64 * 4];   // mask*border-folded bilinear weights
  __shared__ int      sPos[72 * 64];       // (ix<<16) | (iy & 0xffff)
  const int n0 = blockIdx.x * 64, m0 = blockIdx.y * 64, b = blockIdx.z;
  const int tid = threadIdx.x;

  for (int e = tid; e < 72 * 64; e += 128) {
    int gt = e >> 6, n = e & 63;           // gt = g*9+tap == channel in om
    int g = gt / 9, tap = gt - g * 9;
    int hw = n0 + n, y = hw >> 6, x = hw & 63;
    const float* o = om + ((size_t)b * NCM + gt) * HW + hw;
    float py = (float)(y + tap / 3 - 1) + o[0];
    float px = (float)(x + tap % 3 - 1) + o[72 * HW];
    float msk = o[144 * HW];
    float fy0 = floorf(py), fx0 = floorf(px);
    float ly = py - fy0, lx = px - fx0;
    int iy = (int)fy0, ix = (int)fx0;
    float vy0 = (iy >= 0 && iy < Hh) ? 1.f : 0.f;
    float vy1 = (iy + 1 >= 0 && iy + 1 < Hh) ? 1.f : 0.f;
    float vx0 = (ix >= 0 && ix < Wh) ? 1.f : 0.f;
    float vx1 = (ix + 1 >= 0 && ix + 1 < Wh) ? 1.f : 0.f;
    sWgt[e * 4 + 0] = (_Float16)((1.f - ly) * (1.f - lx) * vy0 * vx0 * msk);
    sWgt[e * 4 + 1] = (_Float16)((1.f - ly) * lx * vy0 * vx1 * msk);
    sWgt[e * 4 + 2] = (_Float16)(ly * (1.f - lx) * vy1 * vx0 * msk);
    sWgt[e * 4 + 3] = (_Float16)(ly * lx * vy1 * vx1 * msk);
    sPos[e] = (ix << 16) | (iy & 0xffff);
  }
  __syncthreads();                         // cache ready before gemm prefetches

  const _Float16* wgt = sWgt;
  const int*      pos = sPos;
  gemm64x64(9 * CC,
      [=](int m, int k) {  // w_dcn flat [o][gc][tap] == k = gc*9+tap ordering
        return w_dcn[(size_t)(m0 + m) * (9 * CC) + k];
      },
      [=](int n, int k) {
        int gc = k / 9, tap = k - gc * 9;
        int g = gc >> 4;
        int e = (g * 9 + tap) * 64 + n;
        int p = pos[e];
        int iy = (int)(short)(p & 0xffff);
        int ix = p >> 16;
        int r0 = min(max(iy, 0), Hh - 1) * Wh;
        int r1 = min(max(iy + 1, 0), Hh - 1) * Wh;
        int c0 = min(max(ix, 0), Wh - 1);
        int c1 = min(max(ix + 1, 0), Wh - 1);
        const float* ch = feat_up + ((size_t)b * CC + gc) * HW;
        return (float)wgt[e * 4 + 0] * ch[r0 + c0] +
               (float)wgt[e * 4 + 1] * ch[r0 + c1] +
               (float)wgt[e * 4 + 2] * ch[r1 + c0] +
               (float)wgt[e * 4 + 3] * ch[r1 + c1];
      },
      [=](int m, int n, float v) {
        float r = v + b_dcn[m0 + m];
        r = r > 0.f ? r : 0.f;
        out[((size_t)b * CC + m0 + m) * HW + n0 + n] =
            r + feat_arm[((size_t)b * CC + m0 + m) * HW + n0 + n];
      });
}

// ---------------------------------------------------------------------------
extern "C" void kernel_launch(void* const* d_in, const int* in_sizes, int n_in,
                              void* d_out, int out_size, void* d_ws, size_t ws_size,
                              hipStream_t stream) {
  const float* feat_l  = (const float*)d_in[0];
  const float* feat_s  = (const float*)d_in[1];
  const float* w_atten = (const float*)d_in[2];
  const float* w_fsm   = (const float*)d_in[3];
  const float* w_off   = (const float*)d_in[4];
  const float* w_cm    = (const float*)d_in[5];
  const float* b_cm    = (const float*)d_in[6];
  const float* w_dcn   = (const float*)d_in[7];
  const float* b_dcn   = (const float*)d_in[8];
  float* out = (float*)d_out;

  float* ws       = (float*)d_ws;
  float* feat_up  = ws;                  // 4,194,304 f32
  float* feat_arm = ws + 4194304;        // 4,194,304 f32
  float* off_feat = ws + 8388608;        // 4,194,304 f32
  float* om       = ws + 12582912;       // 7,077,888 f32
  float* scale    = ws + 19660800;       // 1,024 f32

  upsample_kernel<<<(BB * CC * HW + 255) / 256, 256, 0, stream>>>(feat_s, feat_up);
  gap_atten_kernel<<<BB, 256, 0, stream>>>(feat_l, w_atten, scale);
  fsm_kernel<<<dim3(64, 2, BB), 128, 0, stream>>>(feat_l, w_fsm, scale, feat_arm);
  off_kernel<<<dim3(64, 2, BB), 128, 0, stream>>>(feat_arm, feat_up, w_off, off_feat);
  cm_kernel<<<dim3(64, 4, BB), 128, 0, stream>>>(off_feat, w_cm, b_cm, om);
  dcn_kernel<<<dim3(64, 2, BB), 128, 0, stream>>>(feat_up, om, w_dcn, b_dcn, feat_arm, out);
}
